// GAT_12189117186499
// MI455X (gfx1250) — matrix operand order repaired and checked by reading, not compiled, not run on hardware
//
#include <hip/hip_runtime.h>
#include <hip/hip_bf16.h>

typedef __attribute__((ext_vector_type(16))) _Float16 v16h;
typedef __attribute__((ext_vector_type(8)))  _Float16 v8h;
typedef __attribute__((ext_vector_type(8)))  float    v8f;

#define N_NODES  20000
#define N_EDGES  640000
#define N_GRAPHS 256
#define D_IN     7
#define L_DIM    16
#define H_OUT    64

// ---------------------------------------------------------------------------
// ordered-int encoding for float atomic max
// ---------------------------------------------------------------------------
__device__ __forceinline__ int enc_ord(float f) {
  int i = __float_as_int(f);
  return (i >= 0) ? i : (i ^ 0x7fffffff);
}
__device__ __forceinline__ float dec_ord(int i) {
  return __int_as_float((i >= 0) ? i : (i ^ 0x7fffffff));
}
#define ENC_NEG_INF 0x807fffff   // enc_ord(-inf)

// ---------------------------------------------------------------------------
// operand staging: f32 -> f16 with zero pad to Kp (row-major [rows,Kp])
// ---------------------------------------------------------------------------
__global__ void cvt_pad_f16(const float* __restrict__ src, _Float16* __restrict__ dst,
                            int rows, int K, int Kp) {
  int i = blockIdx.x * blockDim.x + threadIdx.x;
  if (i >= rows * Kp) return;
  int r = i / Kp, k = i - r * Kp;
  dst[i] = (k < K) ? (_Float16)src[r * K + k] : (_Float16)0.f;
}
// W[K,Nf] f32 -> Wt[Nf,Kp] f16 (transpose + pad)
__global__ void cvt_w_f16t(const float* __restrict__ W, _Float16* __restrict__ Wt,
                           int K, int Kp, int Nf) {
  int i = blockIdx.x * blockDim.x + threadIdx.x;
  if (i >= Nf * Kp) return;
  int n = i / Kp, k = i - n * Kp;
  Wt[i] = (k < K) ? (_Float16)W[k * Nf + n] : (_Float16)0.f;
}

// ---------------------------------------------------------------------------
// WMMA GEMM: C[M,Nf] = A16[M,Kp] * Wt16[Nf,Kp]^T   (f16 in, f32 accum/out)
// grid = (M/16, Nf/16), block = 32 (one wave per 16x16 tile). Kp % 32 == 0.
// Branch-free inner loop: 4x b128 loads + 1 wmma per 32-K step.
// ---------------------------------------------------------------------------
__global__ void gat_gemm_wmma(const _Float16* __restrict__ A16,
                              const _Float16* __restrict__ Bt16,
                              float* __restrict__ C, int M, int Kp, int Nf) {
  const int lane = threadIdx.x & 31;
  const int row0 = blockIdx.x << 4;
  const int col0 = blockIdx.y << 4;
  const int g = lane >> 4;      // K half-group select
  const int l = lane & 15;
  const _Float16* __restrict__ Arow = A16 + (long)(row0 + l) * Kp;   // A row
  const _Float16* __restrict__ Bcol = Bt16 + (long)(col0 + l) * Kp;  // B column
  v8f acc = {};
  for (int kk = 0; kk < Kp; kk += 32) {
    // A fragment: halves 0-7 -> K = kk+g*8+0..7 ; halves 8-15 -> +16
    const int ka = kk + g * 8;
    v8h a0 = *(const v8h*)(Arow + ka);
    v8h a1 = *(const v8h*)(Arow + ka + 16);
    // B fragment: halves 0-7 -> K = kk+g*8+0..7 ; halves 8-15 -> +16
    const int kb = kk + g * 8;
    v8h b0 = *(const v8h*)(Bcol + kb);
    v8h b1 = *(const v8h*)(Bcol + kb + 16);
    v16h a, b;
    #pragma unroll
    for (int i = 0; i < 8; ++i) {
      a[i] = a0[i]; a[i + 8] = a1[i];
      b[i] = b0[i]; b[i + 8] = b1[i];
    }
    acc = __builtin_amdgcn_wmma_f32_16x16x32_f16(false, a, false, b,
                                                 (short)0, acc, false, false);
  }
  // C/D layout: VGPR r -> row r (lanes 0-15) / row r+8 (lanes 16-31), col = lane&15
  #pragma unroll
  for (int r = 0; r < 8; ++r)
    C[(long)(row0 + r + 8 * g) * Nf + (col0 + l)] = acc[r];
}

// ---------------------------------------------------------------------------
// per-node attention scalars: a_s[n] = h[n,:]·as, a_d[n] = h[n,:]·ad
// one wave per node
// ---------------------------------------------------------------------------
__global__ void gat_attn_coeffs(const float* __restrict__ h,
                                const float* __restrict__ asv,
                                const float* __restrict__ adv,
                                float* __restrict__ a_s, float* __restrict__ a_d,
                                int N, int F) {
  int wave = (int)((blockIdx.x * blockDim.x + threadIdx.x) >> 5);
  int lane = threadIdx.x & 31;
  if (wave >= N) return;
  const float* row = h + (long)wave * F;
  float s = 0.f, d = 0.f;
  for (int f = lane; f < F; f += 32) {
    float v = row[f];
    s += v * asv[f];
    d += v * adv[f];
  }
  #pragma unroll
  for (int off = 16; off; off >>= 1) {
    s += __shfl_xor(s, off, 32);
    d += __shfl_xor(d, off, 32);
  }
  if (lane == 0) { a_s[wave] = s; a_d[wave] = d; }
}

// ---------------------------------------------------------------------------
// edge logits + segment-max (self loops appended as edges E..E+N-1)
// ---------------------------------------------------------------------------
__global__ void gat_edge_logits(const int* __restrict__ src,
                                const int* __restrict__ dst,
                                const float* __restrict__ ecol,
                                const float* __restrict__ a_s,
                                const float* __restrict__ a_d,
                                const float* __restrict__ scal, // [0]=ea_mean [1]=coef
                                float* __restrict__ logits,
                                int* __restrict__ menc, int E, int N) {
  int e = blockIdx.x * blockDim.x + threadIdx.x;
  if (e >= E + N) return;
  int s, d; float ea;
  if (e < E) { s = src[e]; d = dst[e]; ea = ecol[e]; }
  else       { s = d = e - E;          ea = scal[0]; }
  float x = a_s[s] + a_d[d] + ea * scal[1];
  x = (x > 0.f) ? x : 0.2f * x;          // leaky_relu(0.2)
  logits[e] = x;
  atomicMax(&menc[d], enc_ord(x));
}

// exp(logit - max) + segment-sum of denominators
__global__ void gat_edge_exp(const int* __restrict__ dst,
                             float* __restrict__ logits,
                             const int* __restrict__ menc,
                             float* __restrict__ denom, int E, int N) {
  int e = blockIdx.x * blockDim.x + threadIdx.x;
  if (e >= E + N) return;
  int d = (e < E) ? dst[e] : (e - E);
  float ex = __expf(logits[e] - dec_ord(menc[d]));
  logits[e] = ex;
  atomicAdd(&denom[d], ex);
}

// out[dst,:] += (ex/denom[dst]) * h[src,:]   -- one wave per edge
__global__ void gat_aggregate(const int* __restrict__ src,
                              const int* __restrict__ dst,
                              const float* __restrict__ ex,
                              const float* __restrict__ denom,
                              const float* __restrict__ h,
                              float* __restrict__ out, int E, int N, int F) {
  long t = (long)blockIdx.x * blockDim.x + threadIdx.x;
  int wave = (int)(t >> 5);
  int lane = (int)(t & 31);
  if (wave >= E + N) return;
  int s, d;
  if (wave < E) { s = src[wave]; d = dst[wave]; } else { s = d = wave - E; }
  float alpha = ex[wave] / denom[d];
  const float* hr = h + (long)s * F;
  float* orow = out + (long)d * F;
  for (int f = lane; f < F; f += 32) atomicAdd(&orow[f], alpha * hr[f]);
}

// x = elu(x + b[f]),  F power of two (mask = F-1)
__global__ void gat_bias_elu(float* __restrict__ x, const float* __restrict__ b,
                             int n, int fmask) {
  int i = blockIdx.x * blockDim.x + threadIdx.x;
  if (i >= n) return;
  float v = x[i] + b[i & fmask];
  x[i] = (v > 0.f) ? v : (__expf(v) - 1.f);
}

// ---------------------------------------------------------------------------
// small utility kernels
// ---------------------------------------------------------------------------
__global__ void fill_f32(float* p, float v, int n) {
  int i = blockIdx.x * blockDim.x + threadIdx.x;
  if (i < n) p[i] = v;
}
__global__ void fill_i32(int* p, int v, int n) {
  int i = blockIdx.x * blockDim.x + threadIdx.x;
  if (i < n) p[i] = v;
}
// single-block mean of n floats -> *outp
__global__ void reduce_mean(const float* __restrict__ x, int n, float* outp) {
  __shared__ float sm[256];
  float s = 0.f;
  for (int i = threadIdx.x; i < n; i += 256) s += x[i];
  sm[threadIdx.x] = s; __syncthreads();
  for (int st = 128; st; st >>= 1) {
    if ((int)threadIdx.x < st) sm[threadIdx.x] += sm[threadIdx.x + st];
    __syncthreads();
  }
  if (threadIdx.x == 0) *outp = sm[0] / (float)n;
}
// single-block dot(a,b) of n floats -> *outp
__global__ void dot_small(const float* __restrict__ a, const float* __restrict__ b,
                          int n, float* outp) {
  __shared__ float sm[256];
  float s = 0.f;
  for (int i = threadIdx.x; i < n; i += 256) s += a[i] * b[i];
  sm[threadIdx.x] = s; __syncthreads();
  for (int st = 128; st; st >>= 1) {
    if ((int)threadIdx.x < st) sm[threadIdx.x] += sm[threadIdx.x + st];
    __syncthreads();
  }
  if (threadIdx.x == 0) *outp = sm[0];
}
// scatter-add pooling
__global__ void pool_sum(const float* __restrict__ h, const int* __restrict__ batch,
                         float* __restrict__ gsum, float* __restrict__ gcnt,
                         int N, int F) {
  int i = blockIdx.x * blockDim.x + threadIdx.x;
  if (i >= N * F) return;
  int n = i / F, f = i - n * F;
  int b = batch[n];
  atomicAdd(&gsum[b * F + f], h[i]);
  if (f == 0) atomicAdd(&gcnt[b], 1.f);
}
// o[g,j] = (gsum[g,:]/cnt ++ xn[g,:]) @ linW + linb
__global__ void final_linear(const float* __restrict__ gsum,
                             const float* __restrict__ gcnt,
                             const float* __restrict__ xn,
                             const float* __restrict__ linW,
                             const float* __restrict__ linb,
                             float* __restrict__ out, int G, int H, int Ld, int O) {
  int i = blockIdx.x * blockDim.x + threadIdx.x;
  if (i >= G * O) return;
  int g = i / O, j = i - g * O;
  float cnt = gcnt[g]; if (cnt < 1.f) cnt = 1.f;
  float s = linb[j];
  float inv = 1.f / cnt;
  for (int k = 0; k < H; ++k)  s += gsum[g * H + k] * inv * linW[k * O + j];
  for (int l = 0; l < Ld; ++l) s += xn[g * Ld + l] * linW[(H + l) * O + j];
  out[i] = s;
}

// ---------------------------------------------------------------------------
// host orchestration
// ---------------------------------------------------------------------------
static inline int ceil_div(long a, int b) { return (int)((a + b - 1) / b); }

extern "C" void kernel_launch(void* const* d_in, const int* in_sizes, int n_in,
                              void* d_out, int out_size, void* d_ws, size_t ws_size,
                              hipStream_t stream) {
  (void)in_sizes; (void)n_in; (void)out_size; (void)ws_size;

  const float* x_in[2]   = { (const float*)d_in[0], (const float*)d_in[1] };
  const int*   eidx[2]   = { (const int*)d_in[2],   (const int*)d_in[3]   };
  const int*   batch[2]  = { (const int*)d_in[4],   (const int*)d_in[5]   };
  const float* xn2[2]    = { (const float*)d_in[6], (const float*)d_in[7] };
  const float* ecol[2]   = { (const float*)d_in[8], (const float*)d_in[9] };
  const float *W[3], *asv[3], *adv[3], *Wev[3], *aev[3], *bv[3];
  for (int i = 0; i < 3; ++i) {
    int base = 10 + 6 * i;
    W[i]   = (const float*)d_in[base + 0];
    asv[i] = (const float*)d_in[base + 1];
    adv[i] = (const float*)d_in[base + 2];
    Wev[i] = (const float*)d_in[base + 3];
    aev[i] = (const float*)d_in[base + 4];
    bv[i]  = (const float*)d_in[base + 5];
  }
  const float* linW = (const float*)d_in[28];
  const float* linb = (const float*)d_in[29];
  float* out = (float*)d_out;

  // workspace layout (all offsets 64B aligned) ~28.7 MB
  char* ws = (char*)d_ws;
  float*    bufX  = (float*)   (ws + 0);           // node features / aggregation out
  float*    bufH  = (float*)   (ws + 10240000);    // h = x@W
  _Float16* xh    = (_Float16*)(ws + 20480000);    // f16 padded x (N x 128)
  _Float16* Wt    = (_Float16*)(ws + 25600000);    // f16 transposed W (128 x 128)
  float*    a_s   = (float*)   (ws + 25632768);
  float*    a_d   = (float*)   (ws + 25712768);
  int*      menc  = (int*)     (ws + 25792768);
  float*    denom = (float*)   (ws + 25872768);
  float*    logi  = (float*)   (ws + 25952768);    // E+N floats
  float*    scal  = (float*)   (ws + 28592768);    // [0]=ea_mean [1]=We·ae
  float*    gsum  = (float*)   (ws + 28592832);    // 256*64
  float*    gcnt  = (float*)   (ws + 28658368);    // 256

  const int Kdims[3]  = { D_IN, 128, 128 };
  const int Kpdims[3] = { 32,   128, 128 };   // K padded to multiple of 32
  const int Fdims[3]  = { 128,  128, H_OUT };
  const int ET = N_EDGES + N_NODES;

  for (int br = 0; br < 2; ++br) {
    const int* src = eidx[br];
    const int* dst = eidx[br] + N_EDGES;

    reduce_mean<<<1, 256, 0, stream>>>(ecol[br], N_EDGES, &scal[0]);

    const float* xcur = x_in[br];
    for (int li = 0; li < 3; ++li) {
      const int K = Kdims[li], Kp = Kpdims[li], F = Fdims[li];

      dot_small<<<1, 256, 0, stream>>>(Wev[li], aev[li], F, &scal[1]);

      // stage operands in f16 (pad x rows to Kp; transpose+pad W)
      cvt_pad_f16<<<ceil_div((long)N_NODES * Kp, 256), 256, 0, stream>>>(
          xcur, xh, N_NODES, K, Kp);
      cvt_w_f16t<<<ceil_div((long)F * Kp, 256), 256, 0, stream>>>(
          W[li], Wt, K, Kp, F);

      // h = x @ W  (WMMA, branch-free b128 inner loop)
      dim3 gg(N_NODES / 16, F / 16);
      gat_gemm_wmma<<<gg, 32, 0, stream>>>(xh, Wt, bufH, N_NODES, Kp, F);

      gat_attn_coeffs<<<ceil_div((long)N_NODES * 32, 256), 256, 0, stream>>>(
          bufH, asv[li], adv[li], a_s, a_d, N_NODES, F);

      fill_i32<<<ceil_div(N_NODES, 256), 256, 0, stream>>>(menc, (int)ENC_NEG_INF, N_NODES);
      fill_f32<<<ceil_div(N_NODES, 256), 256, 0, stream>>>(denom, 0.f, N_NODES);
      fill_f32<<<ceil_div((long)N_NODES * F, 256), 256, 0, stream>>>(bufX, 0.f, N_NODES * F);

      gat_edge_logits<<<ceil_div(ET, 256), 256, 0, stream>>>(
          src, dst, ecol[br], a_s, a_d, scal, logi, menc, N_EDGES, N_NODES);
      gat_edge_exp<<<ceil_div(ET, 256), 256, 0, stream>>>(
          dst, logi, menc, denom, N_EDGES, N_NODES);
      gat_aggregate<<<ceil_div((long)ET * 32, 256), 256, 0, stream>>>(
          src, dst, logi, denom, bufH, bufX, N_EDGES, N_NODES, F);

      gat_bias_elu<<<ceil_div((long)N_NODES * F, 256), 256, 0, stream>>>(
          bufX, bv[li], N_NODES * F, F - 1);

      xcur = bufX;
    }

    // mean pool + final linear
    fill_f32<<<ceil_div(N_GRAPHS * H_OUT, 256), 256, 0, stream>>>(gsum, 0.f, N_GRAPHS * H_OUT);
    fill_f32<<<1, 256, 0, stream>>>(gcnt, 0.f, N_GRAPHS);
    pool_sum<<<ceil_div((long)N_NODES * H_OUT, 256), 256, 0, stream>>>(
        bufX, batch[br], gsum, gcnt, N_NODES, H_OUT);
    final_linear<<<ceil_div(N_GRAPHS * H_OUT, 256), 256, 0, stream>>>(
        gsum, gcnt, xn2[br], linW, linb, out + br * N_GRAPHS * H_OUT,
        N_GRAPHS, H_OUT, L_DIM, H_OUT);
  }
}
